// GRU_76020921139952
// MI455X (gfx1250) — matrix operand order repaired and checked by reading, not compiled, not run on hardware
//
#include <hip/hip_runtime.h>

// GRU: B=64, T=512, I=1024, H=1024  (fp32 in/out, bf16 WMMA with f32 accum)
#define BB 64
#define TT 512
#define II 1024
#define HH 1024

typedef __attribute__((ext_vector_type(16))) __bf16 v16bf;
typedef __attribute__((ext_vector_type(8)))  float  v8f;

static __device__ __forceinline__ unsigned short f2bf(float f) {
    unsigned u = __float_as_uint(f);
    u += 0x7FFFu + ((u >> 16) & 1u);          // round-to-nearest-even
    return (unsigned short)(u >> 16);
}

static __device__ __forceinline__ float sigm(float x) {
    return 1.0f / (1.0f + __expf(-x));
}

// A-tile (16x32 bf16, M x K):  lanes 0-15 row M=lane, K = {k0..k0+7, k0+16..k0+23}
//                              lanes 16-31 row M=lane-16, K = {k0+8.., k0+24..}
static __device__ __forceinline__ v16bf ld_a(const unsigned short* __restrict__ mat,
                                             int ld, int k0) {
    const int lane = threadIdx.x & 31;
    const unsigned short* p = mat + (size_t)(lane & 15) * ld + k0 + ((lane & 16) ? 8 : 0);
    union { uint4 q[2]; v16bf v; } u;
    u.q[0] = *(const uint4*)(p);
    u.q[1] = *(const uint4*)(p + 16);
    return u.v;
}

// B-tile (32x16 bf16, K x N): lane = output column n (lanes 0-15 K={k0..k0+7, k0+16..k0+23},
// lanes 16-31 K={k0+8.., k0+24..}).  Column n of B == row n of the row-major weight
// matrix (we compute x @ W^T), so each lane reads two 16B runs.
static __device__ __forceinline__ v16bf ld_b(const unsigned short* __restrict__ wrow0,
                                             int ld, int k0) {
    const int lane = threadIdx.x & 31;
    const unsigned short* p = wrow0 + (size_t)(lane & 15) * ld + k0 + ((lane & 16) ? 8 : 0);
    union { uint4 q[2]; v16bf v; } u;
    u.q[0] = *(const uint4*)(p);
    u.q[1] = *(const uint4*)(p + 16);
    return u.v;
}

static __device__ __forceinline__ v8f wmma_bf16(v16bf a, v16bf b, v8f c) {
    return __builtin_amdgcn_wmma_f32_16x16x32_bf16(false, a, false, b,
                                                   (short)0, c, false, false);
}

// ---------------------------------------------------------------- converters
__global__ void f2bf_kernel(const float* __restrict__ in,
                            unsigned short* __restrict__ out, size_t n) {
    size_t i = (size_t)blockIdx.x * blockDim.x + threadIdx.x;
    if (i < n) out[i] = f2bf(in[i]);
}

__global__ void init_h_kernel(const float* __restrict__ h0,
                              float* __restrict__ h32,
                              unsigned short* __restrict__ h16, int n) {
    int i = blockIdx.x * blockDim.x + threadIdx.x;
    if (i < n) { float v = h0[i]; h32[i] = v; h16[i] = f2bf(v); }
}

// ------------------------------------------------- phase 1: input projections
// X[(b*T+t), I] @ Wg[n, I]^T + bias -> xproj[g][(b*T+t), H]
// block = 256 thr (8 waves); wave owns 16 rows of M; block owns 64 cols of N.
// Software-pipelined: next k-tile loads issued before current WMMAs so global
// latency overlaps matrix issue instead of s_wait_loadcnt 0 serializing them.
__global__ __launch_bounds__(256)
void xproj_gemm(const unsigned short* __restrict__ X,
                const unsigned short* __restrict__ Wi,
                const float* __restrict__ b0, const float* __restrict__ b1,
                const float* __restrict__ b2, float* __restrict__ out) {
    const int lane = threadIdx.x & 31;
    const int wave = threadIdx.x >> 5;
    const int m0 = blockIdx.x * 128 + wave * 16;
    const int n0 = blockIdx.y * 64;
    const int g  = blockIdx.z;

    const unsigned short* A = X + (size_t)m0 * II;
    const unsigned short* W = Wi + (size_t)g * HH * II;
    const unsigned short* W0 = W + (size_t)(n0 +  0) * II;
    const unsigned short* W1 = W + (size_t)(n0 + 16) * II;
    const unsigned short* W2 = W + (size_t)(n0 + 32) * II;
    const unsigned short* W3 = W + (size_t)(n0 + 48) * II;
    const float* bias = (g == 0) ? b0 : (g == 1) ? b1 : b2;
    float* C = out + (size_t)g * ((size_t)BB * TT * HH);

    v8f acc[4] = {};

    // prologue: tile 0
    v16bf a  = ld_a(A,  II, 0);
    v16bf w0 = ld_b(W0, II, 0);
    v16bf w1 = ld_b(W1, II, 0);
    v16bf w2 = ld_b(W2, II, 0);
    v16bf w3 = ld_b(W3, II, 0);

#pragma unroll 2
    for (int k0 = 0; k0 < II; k0 += 32) {
        const int kn = (k0 + 32 < II) ? (k0 + 32) : 0;   // harmless reload on last iter
        // prefetch next tiles (loads in flight during WMMAs below)
        v16bf an  = ld_a(A,  II, kn);
        v16bf wn0 = ld_b(W0, II, kn);
        v16bf wn1 = ld_b(W1, II, kn);
        v16bf wn2 = ld_b(W2, II, kn);
        v16bf wn3 = ld_b(W3, II, kn);

        acc[0] = wmma_bf16(a, w0, acc[0]);
        acc[1] = wmma_bf16(a, w1, acc[1]);
        acc[2] = wmma_bf16(a, w2, acc[2]);
        acc[3] = wmma_bf16(a, w3, acc[3]);

        a = an; w0 = wn0; w1 = wn1; w2 = wn2; w3 = wn3;
    }

    const int mrow = m0 + ((lane & 16) ? 8 : 0);
#pragma unroll
    for (int j = 0; j < 4; ++j) {
        const int n = n0 + j * 16 + (lane & 15);
        const float bv = bias[n];
#pragma unroll
        for (int i = 0; i < 8; ++i)
            C[(size_t)(mrow + i) * HH + n] = acc[j][i] + bv;
    }
}

// --------------------------------------------------- phase 2: recurrent step
// block = 128 thr (4 waves); wave owns 16 batch rows; block owns 16 cols of H,
// computing all 3 recurrent gates + fused GRU cell update. Software-pipelined.
__global__ __launch_bounds__(128, 4)
void gru_step(const unsigned short* __restrict__ h16,
              const float* __restrict__ h32,
              const unsigned short* __restrict__ Wh,   // [3, H, H] bf16
              const float* __restrict__ bhr, const float* __restrict__ bhz,
              const float* __restrict__ bhn,
              const float* __restrict__ xproj,         // [3, B*T, H] f32
              int t,
              float* __restrict__ outs,                // [B, T, H]
              float* __restrict__ h32n,
              unsigned short* __restrict__ h16n,
              float* __restrict__ hlast) {
    const int lane = threadIdx.x & 31;
    const int wave = threadIdx.x >> 5;
    const int m0 = wave * 16;            // batch rows
    const int n0 = blockIdx.x * 16;      // hidden cols

    const unsigned short* A  = h16 + (size_t)m0 * HH;
    const unsigned short* Wr = Wh + (size_t)0 * HH * HH + (size_t)n0 * HH;
    const unsigned short* Wz = Wh + (size_t)1 * HH * HH + (size_t)n0 * HH;
    const unsigned short* Wn = Wh + (size_t)2 * HH * HH + (size_t)n0 * HH;

    v8f ar = {}, az = {}, an_ = {};

    v16bf a  = ld_a(A,  HH, 0);
    v16bf br_ = ld_b(Wr, HH, 0);
    v16bf bz_ = ld_b(Wz, HH, 0);
    v16bf bn_ = ld_b(Wn, HH, 0);

#pragma unroll 2
    for (int k0 = 0; k0 < HH; k0 += 32) {
        const int kn = (k0 + 32 < HH) ? (k0 + 32) : 0;
        v16bf a2  = ld_a(A,  HH, kn);
        v16bf br2 = ld_b(Wr, HH, kn);
        v16bf bz2 = ld_b(Wz, HH, kn);
        v16bf bn2 = ld_b(Wn, HH, kn);

        ar  = wmma_bf16(a, br_, ar);
        az  = wmma_bf16(a, bz_, az);
        an_ = wmma_bf16(a, bn_, an_);

        a = a2; br_ = br2; bz_ = bz2; bn_ = bn2;
    }

    const int n = n0 + (lane & 15);
    const float cbr = bhr[n], cbz = bhz[n], cbn = bhn[n];
    const int mbase = m0 + ((lane & 16) ? 8 : 0);
    const size_t BTH = (size_t)BB * TT * HH;
    const float* xr = xproj;
    const float* xz = xproj + BTH;
    const float* xn = xproj + 2 * BTH;

#pragma unroll
    for (int i = 0; i < 8; ++i) {
        const int b = mbase + i;
        const size_t xi = ((size_t)b * TT + t) * HH + n;
        const size_t hi = (size_t)b * HH + n;
        float rv = sigm(xr[xi] + ar[i] + cbr);
        float zv = sigm(xz[xi] + az[i] + cbz);
        float nv = tanhf(xn[xi] + rv * (an_[i] + cbn));
        float hv = (1.0f - zv) * nv + zv * h32[hi];
        outs[xi] = hv;
        h32n[hi] = hv;
        h16n[hi] = f2bf(hv);
        if (hlast) hlast[hi] = hv;
    }
}

// ---------------------------------------------------------------------- host
extern "C" void kernel_launch(void* const* d_in, const int* in_sizes, int n_in,
                              void* d_out, int out_size, void* d_ws, size_t ws_size,
                              hipStream_t stream) {
    const float* inputs = (const float*)d_in[0];
    const float* h0     = (const float*)d_in[1];
    const float* w_ir = (const float*)d_in[2];
    const float* w_iz = (const float*)d_in[3];
    const float* w_in = (const float*)d_in[4];
    const float* b_ir = (const float*)d_in[5];
    const float* b_iz = (const float*)d_in[6];
    const float* b_in = (const float*)d_in[7];
    const float* w_hr = (const float*)d_in[8];
    const float* w_hz = (const float*)d_in[9];
    const float* w_hn = (const float*)d_in[10];
    const float* b_hr = (const float*)d_in[11];
    const float* b_hz = (const float*)d_in[12];
    const float* b_hn = (const float*)d_in[13];

    const size_t BTI = (size_t)BB * TT * II;
    const size_t BTH = (size_t)BB * TT * HH;
    const size_t HI  = (size_t)HH * II;
    const size_t BH  = (size_t)BB * HH;

    // workspace carve-out
    char* p = (char*)d_ws;
    unsigned short* X16  = (unsigned short*)p;  p += BTI * 2;          // 64 MB
    unsigned short* Wi16 = (unsigned short*)p;  p += 3 * HI * 2;       // 6 MB
    unsigned short* Wh16 = (unsigned short*)p;  p += 3 * HI * 2;       // 6 MB
    float*          Xpr  = (float*)p;           p += 3 * BTH * 4;      // 402 MB
    float*          hA32 = (float*)p;           p += BH * 4;
    float*          hB32 = (float*)p;           p += BH * 4;
    unsigned short* hA16 = (unsigned short*)p;  p += BH * 2;
    unsigned short* hB16 = (unsigned short*)p;  p += BH * 2;
    (void)ws_size; (void)n_in; (void)in_sizes; (void)out_size;

    // --- bf16 conversions
    f2bf_kernel<<<dim3((unsigned)((BTI + 255) / 256)), 256, 0, stream>>>(inputs, X16, BTI);
    f2bf_kernel<<<dim3((unsigned)((HI + 255) / 256)), 256, 0, stream>>>(w_ir, Wi16 + 0 * HI, HI);
    f2bf_kernel<<<dim3((unsigned)((HI + 255) / 256)), 256, 0, stream>>>(w_iz, Wi16 + 1 * HI, HI);
    f2bf_kernel<<<dim3((unsigned)((HI + 255) / 256)), 256, 0, stream>>>(w_in, Wi16 + 2 * HI, HI);
    f2bf_kernel<<<dim3((unsigned)((HI + 255) / 256)), 256, 0, stream>>>(w_hr, Wh16 + 0 * HI, HI);
    f2bf_kernel<<<dim3((unsigned)((HI + 255) / 256)), 256, 0, stream>>>(w_hz, Wh16 + 1 * HI, HI);
    f2bf_kernel<<<dim3((unsigned)((HI + 255) / 256)), 256, 0, stream>>>(w_hn, Wh16 + 2 * HI, HI);
    init_h_kernel<<<dim3((unsigned)((BH + 255) / 256)), 256, 0, stream>>>(h0, hA32, hA16, (int)BH);

    // --- phase 1: batched input projections (WMMA bf16 -> f32)
    {
        dim3 grid((BB * TT) / 128, HH / 64, 3);
        xproj_gemm<<<grid, 256, 0, stream>>>(X16, Wi16, b_ir, b_iz, b_in, Xpr);
    }

    // --- phase 2: sequential scan, ping-pong h buffers
    float*          h32buf[2] = { hA32, hB32 };
    unsigned short* h16buf[2] = { hA16, hB16 };
    float* outs  = (float*)d_out;
    float* hlast = (float*)d_out + BTH;   // tail: [1, B, H]

    for (int t = 0; t < TT; ++t) {
        const int cur = t & 1, nxt = cur ^ 1;
        gru_step<<<dim3(HH / 16), 128, 0, stream>>>(
            h16buf[cur], h32buf[cur], Wh16, b_hr, b_hz, b_hn, Xpr, t,
            outs, h32buf[nxt], h16buf[nxt],
            (t == TT - 1) ? hlast : (float*)nullptr);
    }
}
